// GroupSpecific_43473658970452
// MI455X (gfx1250) — matrix operand order repaired and checked by reading, not compiled, not run on hardware
//
#include <hip/hip_runtime.h>

#define N_SUBNET 4
#define BATCH    16384
#define D_IN     512
#define D_HID    2048
#define D_OUT    512
#define MTILE    64

typedef __attribute__((ext_vector_type(8)))  float  v8f;
typedef __attribute__((ext_vector_type(8)))  __bf16 v8bf;
typedef __attribute__((ext_vector_type(16))) __bf16 v16bf;

static __device__ __forceinline__ v16bf cat16(v8bf a, v8bf b) {
    return __builtin_shufflevector(a, b, 0,1,2,3,4,5,6,7,8,9,10,11,12,13,14,15);
}

// ---------------- routing kernels ----------------
__global__ void hist_k(const int* __restrict__ g, int* __restrict__ counts, int n) {
    int i = blockIdx.x * blockDim.x + threadIdx.x;
    if (i < n) atomicAdd(&counts[g[i]], 1);
}

__global__ void prefix_k(const int* __restrict__ counts, int* __restrict__ offsets) {
    if (threadIdx.x == 0) {
        int s = 0;
        for (int e = 0; e < N_SUBNET; ++e) { offsets[e] = s; s += counts[e]; }
        offsets[N_SUBNET] = s;
    }
}

__global__ void scatter_k(const int* __restrict__ g, const int* __restrict__ offsets,
                          int* __restrict__ cursors, int* __restrict__ perm, int n) {
    int i = blockIdx.x * blockDim.x + threadIdx.x;
    if (i < n) {
        int e = g[i];
        int pos = atomicAdd(&cursors[e], 1);
        perm[offsets[e] + pos] = i;
    }
}

// ---------------- x: f32 -> bf16 straight convert ----------------
__global__ __launch_bounds__(256) void cvt_x_k(const float* __restrict__ in,
                                               __bf16* __restrict__ out, int n) {
    int i = (blockIdx.x * blockDim.x + threadIdx.x) * 4;
    if (i < n) {
        float4 v = *(const float4*)(in + i);
        out[i + 0] = (__bf16)v.x;
        out[i + 1] = (__bf16)v.y;
        out[i + 2] = (__bf16)v.z;
        out[i + 3] = (__bf16)v.w;
    }
}

// ---------------- weight transpose + f32->bf16 convert ----------------
// in: [R][C] f32  ->  out: [C][R] bf16   (per expert in blockIdx.z)
__global__ __launch_bounds__(256) void cvt_t_k(const float* __restrict__ in,
                                               __bf16* __restrict__ out, int R, int C) {
    __shared__ float tile[32][33];
    int gz = blockIdx.z;
    in  += (size_t)gz * R * C;
    out += (size_t)gz * R * C;
    int c0 = blockIdx.x * 32, r0 = blockIdx.y * 32;
    int tx = threadIdx.x, ty = threadIdx.y;   // block (32,8)
    #pragma unroll
    for (int j = 0; j < 32; j += 8)
        tile[ty + j][tx] = in[(size_t)(r0 + ty + j) * C + (c0 + tx)];
    __syncthreads();
    #pragma unroll
    for (int j = 0; j < 32; j += 8)
        out[(size_t)(c0 + ty + j) * R + (r0 + tx)] = (__bf16)tile[tx][ty + j];
}

// ---------------- fused per-expert MLP: async-stage X -> GEMM1+ReLU -> LDS -> GEMM2 ----------------
__global__ __launch_bounds__(256) void moe_mlp_k(
    const __bf16* __restrict__ xbf,    // [BATCH][D_IN] bf16
    const float*  __restrict__ b1,
    const float*  __restrict__ b2,
    const __bf16* __restrict__ w1t,    // [g][D_HID][D_IN] bf16
    const __bf16* __restrict__ w2t,    // [g][D_OUT][D_HID] bf16
    const int*    __restrict__ counts,
    const int*    __restrict__ offsets,
    const int*    __restrict__ perm,
    float*        __restrict__ out)
{
    __shared__ alignas(16) __bf16 hid[MTILE * D_HID];   // 256 KB
    __shared__ alignas(16) __bf16 xa [MTILE * D_IN];    //  64 KB  (320 KB total)

    // map blockIdx.x -> (expert, tile within expert)
    int bid = blockIdx.x;
    int e = -1, tile = 0, base = 0;
    for (int g = 0; g < N_SUBNET; ++g) {
        int t = (counts[g] + MTILE - 1) / MTILE;
        if (bid < base + t) { e = g; tile = bid - base; break; }
        base += t;
    }
    if (e < 0) return;

    int cnt        = counts[e];
    int pbase      = offsets[e] + tile * MTILE;
    int rows_valid = cnt - tile * MTILE;
    if (rows_valid > MTILE) rows_valid = MTILE;

    const __bf16* w1e = w1t + (size_t)e * D_HID * D_IN;
    const __bf16* w2e = w2t + (size_t)e * D_OUT * D_HID;
    const float*  b1e = b1 + e * D_HID;
    const float*  b2e = b2 + e * D_OUT;

    int tid   = threadIdx.x;
    int w     = tid >> 5;        // wave id 0..7
    int lane  = tid & 31;
    int lhalf = lane >> 4;       // 0 | 1
    int l15   = lane & 15;

    // ---- stage gathered X rows into LDS via async global->LDS DMA (ASYNCcnt) ----
    {
        unsigned xa_base = (unsigned)(uintptr_t)&xa[0];
        // 64 rows x 1024 B = 4096 x 16B chunks, 16 per thread
        for (int ch = tid; ch < (MTILE * D_IN * 2) / 16; ch += 256) {
            int row  = ch >> 6;            // 64 chunks per row
            int boff = (ch & 63) * 16;
            int m    = row < rows_valid ? row : 0;
            int tok  = perm[pbase + m];
            const char* gaddr = (const char*)(xbf + (size_t)tok * D_IN) + boff;
            unsigned lds_addr = xa_base + row * (D_IN * 2) + boff;
            asm volatile("global_load_async_to_lds_b128 %0, %1, off"
                         :: "v"(lds_addr), "v"(gaddr) : "memory");
        }
        asm volatile("s_wait_asynccnt 0x0" ::: "memory");
    }
    __syncthreads();

    // ---- GEMM1: [64 x 2048] = X[64 x 512] * W1[512 x 2048], bias + ReLU ----
    // wave w owns hidden columns [w*256, w*256+256)
    for (int nt = 0; nt < 16; ++nt) {
        int n0 = w * 256 + nt * 16;
        float bias = b1e[n0 + l15];
        v8f acc[4];
        #pragma unroll
        for (int mt = 0; mt < 4; ++mt) acc[mt] = (v8f){0,0,0,0,0,0,0,0};
        #pragma unroll 2
        for (int kb = 0; kb < D_IN / 32; ++kb) {
            // B fragment: column n0+l15, K = kb*32 + lhalf*16 + (0..15) contiguous
            const __bf16* brow = w1e + (size_t)(n0 + l15) * D_IN + kb * 32 + lhalf * 16;
            v16bf B = cat16(*(const v8bf*)brow, *(const v8bf*)(brow + 8));
            int ko = kb * 32 + lhalf * 16;  // A: chunks at +0/+8, lane-half -> +16
            #pragma unroll
            for (int mt = 0; mt < 4; ++mt) {
                const __bf16* ar = &xa[(mt * 16 + l15) * D_IN + ko];
                v16bf A = cat16(*(const v8bf*)ar, *(const v8bf*)(ar + 8));
                acc[mt] = __builtin_amdgcn_wmma_f32_16x16x32_bf16(
                              false, A, false, B, (short)0, acc[mt], false, false);
            }
        }
        #pragma unroll
        for (int mt = 0; mt < 4; ++mt) {
            #pragma unroll
            for (int r = 0; r < 8; ++r) {
                int m = mt * 16 + r + lhalf * 8;   // C/D: lanes16-31 hold M=8..15
                float v = acc[mt][r] + bias;
                v = v > 0.f ? v : 0.f;
                hid[m * D_HID + n0 + l15] = (__bf16)v;
            }
        }
    }
    __syncthreads();

    // ---- GEMM2: [64 x 512] = H[64 x 2048] * W2[2048 x 512], + bias, scatter ----
    // wave w owns output columns [w*64, w*64+64)
    for (int nt = 0; nt < 4; ++nt) {
        int n0 = w * 64 + nt * 16;
        float bias = b2e[n0 + l15];
        v8f acc[4];
        #pragma unroll
        for (int mt = 0; mt < 4; ++mt) acc[mt] = (v8f){0,0,0,0,0,0,0,0};
        #pragma unroll 2
        for (int kb = 0; kb < D_HID / 32; ++kb) {
            const __bf16* brow = w2e + (size_t)(n0 + l15) * D_HID + kb * 32 + lhalf * 16;
            v16bf B = cat16(*(const v8bf*)brow, *(const v8bf*)(brow + 8));
            int ko = kb * 32 + lhalf * 16;
            #pragma unroll
            for (int mt = 0; mt < 4; ++mt) {
                const __bf16* ar = &hid[(mt * 16 + l15) * D_HID + ko];
                v16bf A = cat16(*(const v8bf*)ar, *(const v8bf*)(ar + 8));
                acc[mt] = __builtin_amdgcn_wmma_f32_16x16x32_bf16(
                              false, A, false, B, (short)0, acc[mt], false, false);
            }
        }
        #pragma unroll
        for (int mt = 0; mt < 4; ++mt) {
            #pragma unroll
            for (int r = 0; r < 8; ++r) {
                int m = mt * 16 + r + lhalf * 8;
                if (m < rows_valid)
                    out[(size_t)perm[pbase + m] * D_OUT + n0 + l15] = acc[mt][r] + bias;
            }
        }
    }
}

// ---------------- launch ----------------
extern "C" void kernel_launch(void* const* d_in, const int* in_sizes, int n_in,
                              void* d_out, int out_size, void* d_ws, size_t ws_size,
                              hipStream_t stream) {
    const float* x      = (const float*)d_in[0];
    const int*   groups = (const int*)  d_in[1];
    const float* W1     = (const float*)d_in[2];
    const float* b1     = (const float*)d_in[3];
    const float* W2     = (const float*)d_in[4];
    const float* b2     = (const float*)d_in[5];
    float*       out    = (float*)d_out;

    char* ws = (char*)d_ws;
    int* counts  = (int*)ws;        // 4 ints  @ 0
    int* offsets = counts + 4;      // 5 ints  @ 16
    int* cursors = counts + 12;     // 4 ints  @ 48
    int* perm    = counts + 16;     // 16384   @ 64
    __bf16* xbf  = (__bf16*)(ws + 66560);                        // 16 MB
    __bf16* w1t  = xbf + (size_t)BATCH * D_IN;                   //  8 MB
    __bf16* w2t  = w1t + (size_t)N_SUBNET * D_HID * D_IN;        //  8 MB

    hipMemsetAsync(counts, 0, 64, stream);
    hist_k   <<<BATCH / 256, 256, 0, stream>>>(groups, counts, BATCH);
    prefix_k <<<1, 32, 0, stream>>>(counts, offsets);
    scatter_k<<<BATCH / 256, 256, 0, stream>>>(groups, offsets, cursors, perm, BATCH);

    cvt_x_k<<<(BATCH * D_IN / 4) / 256, 256, 0, stream>>>(x, xbf, BATCH * D_IN);

    dim3 tb(32, 8);
    cvt_t_k<<<dim3(D_HID / 32, D_IN  / 32, N_SUBNET), tb, 0, stream>>>(W1, w1t, D_IN,  D_HID);
    cvt_t_k<<<dim3(D_OUT / 32, D_HID / 32, N_SUBNET), tb, 0, stream>>>(W2, w2t, D_HID, D_OUT);

    // max tiles = 256 full + up to 3 partial from 4-way split
    moe_mlp_k<<<BATCH / MTILE + (N_SUBNET - 1), 256, 0, stream>>>(
        xbf, b1, b2, w1t, w2t, counts, offsets, perm, out);
}
